// MultiHeadSelfAttention_61864708931618
// MI455X (gfx1250) — matrix operand order repaired and checked by reading, not compiled, not run on hardware
//
#include <hip/hip_runtime.h>
#include <hip/hip_bf16.h>

// ---------------------------------------------------------------------------
// Multi-head self-attention, bf16 WMMA pipeline for gfx1250 (MI455X).
// B=4, T=2048, C=1024, H=16, D=64.
// GEMMs: 64x64 wave tile (16 WMMA per 8 fragments).
// Attention: transposed-score formulation (S^T = K.Q^T, O^T = V^T.P^T) so the
// softmax is per-lane (1 shuffle per reduction) and alpha-rescales are
// vector x scalar -- minimal VALU/mov overhead around the WMMAs.
// ---------------------------------------------------------------------------

typedef __bf16 bf16_t;
typedef __attribute__((ext_vector_type(16))) __bf16 v16bf;
typedef __attribute__((ext_vector_type(8)))  float  v8f;

#define BATCH 4
#define SEQ   2048
#define CDIM  1024
#define NHEAD 16
#define HDIM  64
#define MROWS (BATCH*SEQ)   // 8192

__device__ __forceinline__ unsigned short bf16_bits(float f) {
  unsigned u = __builtin_bit_cast(unsigned, f);
  unsigned r = u + 0x7FFFu + ((u >> 16) & 1u);   // round-to-nearest-even
  return (unsigned short)(r >> 16);
}
__device__ __forceinline__ bf16_t to_bf16(float f) {
  unsigned short h = bf16_bits(f);
  return __builtin_bit_cast(bf16_t, h);
}
__device__ __forceinline__ unsigned pack_bf16(float lo, float hi) {
  return (unsigned)bf16_bits(lo) | ((unsigned)bf16_bits(hi) << 16);
}

// A-fragment (16x32, MxK): lane holds row m; elems 0..7 = K kbase+8*half..,
// elems 8..15 = K kbase+16+8*half.. (ISA 7.12.2, 16-bit A 16x32)
__device__ __forceinline__ v16bf load_a_frag(const bf16_t* row_ptr, int kbase, int half) {
  v16bf r;
  const bf16_t* p = row_ptr + kbase + 8 * half;
  __builtin_memcpy(&r, p, 16);
  __builtin_memcpy((char*)&r + 16, p + 16, 16);
  return r;
}

// B-fragment (32x16, KxN): lane holds column n; elems 0..7 = K kbase+8*half..,
// elems 8..15 = K kbase+16+8*half..
__device__ __forceinline__ v16bf load_b_frag(const bf16_t* row_ptr, int kbase, int half) {
  v16bf r;
  const bf16_t* p = row_ptr + kbase + 8 * half;
  __builtin_memcpy(&r, p, 16);
  __builtin_memcpy((char*)&r + 16, p + 16, 16);
  return r;
}

#define WMMA_BF16(A, B, C) \
  __builtin_amdgcn_wmma_f32_16x16x32_bf16(false, (A), false, (B), (short)0, (C), false, false)

// ---------------------------------------------------------------------------
// Prep kernels
// ---------------------------------------------------------------------------
__global__ void convert_kernel(const float* __restrict__ in, bf16_t* __restrict__ out, int n) {
  int idx = (blockIdx.x * blockDim.x + threadIdx.x) * 4;
  if (idx + 3 < n) {
    float4 f = *(const float4*)(in + idx);
    out[idx + 0] = to_bf16(f.x);
    out[idx + 1] = to_bf16(f.y);
    out[idx + 2] = to_bf16(f.z);
    out[idx + 3] = to_bf16(f.w);
  }
}

// WT[n][k] = (bf16) W[k][n]   (so WMMA B operand reads contiguous rows)
__global__ void transpose_kernel(const float* __restrict__ W, bf16_t* __restrict__ WT) {
  int idx = blockIdx.x * blockDim.x + threadIdx.x;
  int n = idx >> 10;
  int k = idx & 1023;
  WT[(size_t)n * CDIM + k] = to_bf16(W[(size_t)k * CDIM + n]);
}

// ---------------------------------------------------------------------------
// GEMM: (8192 x 1024) x (1024 x 1024) + bias.
// MODE 0: out = bf16, head-split (B,H,T,D), scaled      (Q and K)
// MODE 1: out = bf16, head-split transposed (B,H,D,T)   (V)
// MODE 2: out = f32, plain row-major (B*T, C)           (final projection)
// Block = 256 thr = 8 waves as 2(M)x4(N); wave tile 64x64; block tile 128x256.
// ---------------------------------------------------------------------------
template <int MODE>
__global__ __launch_bounds__(256) void gemm_kernel(
    const bf16_t* __restrict__ A,    // MROWS x CDIM
    const bf16_t* __restrict__ BTm,  // CDIM x CDIM  (N x K)
    const float*  __restrict__ bias, // CDIM
    void* __restrict__ out,
    float scale)
{
  const int lane = threadIdx.x & 31;
  const int wave = threadIdx.x >> 5;
  const int half = lane >> 4;
  const int l16  = lane & 15;
  const int m0 = blockIdx.y * 128 + (wave >> 2) * 64;
  const int n0 = blockIdx.x * 256 + (wave & 3) * 64;

  v8f c[4][4] = {};
  const bf16_t* arow[4];
  const bf16_t* brow[4];
#pragma unroll
  for (int i = 0; i < 4; i++) {
    arow[i] = A   + (size_t)(m0 + i * 16 + l16) * CDIM;
    brow[i] = BTm + (size_t)(n0 + i * 16 + l16) * CDIM;
  }

  for (int kk = 0; kk < CDIM; kk += 32) {
    v16bf a[4], b[4];
#pragma unroll
    for (int i = 0; i < 4; i++) a[i] = load_a_frag(arow[i], kk, half);
#pragma unroll
    for (int j = 0; j < 4; j++) b[j] = load_b_frag(brow[j], kk, half);
#pragma unroll
    for (int i = 0; i < 4; i++)
#pragma unroll
      for (int j = 0; j < 4; j++)
        c[i][j] = WMMA_BF16(a[i], b[j], c[i][j]);
  }

#pragma unroll
  for (int i = 0; i < 4; i++)
#pragma unroll
    for (int j = 0; j < 4; j++) {
      int ncol = n0 + j * 16 + l16;
      float bv = bias[ncol];
#pragma unroll
      for (int r = 0; r < 8; r++) {
        int mrow = m0 + i * 16 + r + 8 * half;   // C/D layout: M = r + 8*half
        float v = (c[i][j][r] + bv) * scale;
        if (MODE == 2) {
          ((float*)out)[(size_t)mrow * CDIM + ncol] = v;
        } else {
          int b_ = mrow >> 11, t = mrow & (SEQ - 1);
          int h  = ncol >> 6,  d = ncol & (HDIM - 1);
          if (MODE == 0)
            ((bf16_t*)out)[(((size_t)(b_ * NHEAD + h)) * SEQ + t) * HDIM + d] = to_bf16(v);
          else
            ((bf16_t*)out)[(((size_t)(b_ * NHEAD + h)) * HDIM + d) * SEQ + t] = to_bf16(v);
        }
      }
    }
}

// ---------------------------------------------------------------------------
// Flash attention, transposed-score formulation.
// Per wave: 32 queries (two 16-query N-tiles), keys in blocks of 32.
//   S^T tile = K_block(32k x 64d) . Q^T(64d x 16q)  -> C: M=key, N=query=lane
//   => each lane holds 16 of its query's 32 scores; partner lane (xor 16)
//      holds the rest: softmax = in-lane tree + ONE shuffle per reduction.
//   O^T(64d x 16q) += V^T(64d x 32k) . P^T(32k x 16q)
// Q pre-scaled by (1/sqrt(D))*log2(e) so softmax uses exp2 (v_exp_f32).
// ---------------------------------------------------------------------------
__global__ __launch_bounds__(256) void attn_kernel(
    const bf16_t* __restrict__ Q,   // (B*H, T, D), pre-scaled
    const bf16_t* __restrict__ K,   // (B*H, T, D)
    const bf16_t* __restrict__ Vt,  // (B*H, D, T)
    bf16_t* __restrict__ O)         // (B, T, C)
{
  __shared__ bf16_t plds[8][32 * 32];   // 2 KB per wave: P^T rows [query][key]
  const int lane = threadIdx.x & 31;
  const int wave = threadIdx.x >> 5;
  const int half = lane >> 4;
  const int l16  = lane & 15;
  const int bh = blockIdx.y;
  const int q0 = blockIdx.x * 256 + wave * 32;

  const bf16_t* Qp = Q  + ((size_t)bh * SEQ + q0) * HDIM;
  const bf16_t* Kp = K  + (size_t)bh * SEQ * HDIM;
  const bf16_t* Vp = Vt + (size_t)bh * HDIM * SEQ;

  // Persistent Q B-fragments: (d x q), two query tiles x two 32-d slices
  v16bf bq[2][2];
#pragma unroll
  for (int u = 0; u < 2; u++)
#pragma unroll
    for (int ds = 0; ds < 2; ds++)
      bq[u][ds] = load_b_frag(Qp + (size_t)(u * 16 + l16) * HDIM, ds * 32, half);

  v8f o[4][2] = {};            // O^T accum: [dim-tile][query-tile]
  float rmax[2] = {-1e30f, -1e30f};   // per-lane: stats of THIS lane's query
  float rsum[2] = {0.f, 0.f};

  bf16_t* P = plds[wave];

  for (int kb = 0; kb < SEQ; kb += 32) {
    // K A-fragments: two 16-key tiles x two 32-d slices
    v16bf ak[2][2];
#pragma unroll
    for (int kt = 0; kt < 2; kt++)
#pragma unroll
      for (int ds = 0; ds < 2; ds++)
        ak[kt][ds] = load_a_frag(Kp + (size_t)(kb + kt * 16 + l16) * HDIM, ds * 32, half);

#pragma unroll
    for (int u = 0; u < 2; u++) {
      v8f s0 = {}, s1 = {};
      s0 = WMMA_BF16(ak[0][0], bq[u][0], s0);
      s0 = WMMA_BF16(ak[0][1], bq[u][1], s0);
      s1 = WMMA_BF16(ak[1][0], bq[u][0], s1);
      s1 = WMMA_BF16(ak[1][1], bq[u][1], s1);

      // This lane's 16 scores for query (q0+u*16+l16):
      //   s0[r] -> key kb + 8*half + r ;  s1[r] -> key kb + 16 + 8*half + r
      float mloc = fmaxf(s0[0], s1[0]);
#pragma unroll
      for (int r = 1; r < 8; r++) mloc = fmaxf(mloc, fmaxf(s0[r], s1[r]));
      mloc = fmaxf(mloc, __shfl_xor(mloc, 16, 32));       // other 16 keys
      float mnew  = fmaxf(rmax[u], mloc);
      float alpha = exp2f(rmax[u] - mnew);

      float p0[8], p1[8], ssum = 0.f;
#pragma unroll
      for (int r = 0; r < 8; r++) {
        p0[r] = exp2f(s0[r] - mnew);
        p1[r] = exp2f(s1[r] - mnew);
        ssum += p0[r] + p1[r];
      }
      ssum += __shfl_xor(ssum, 16, 32);
      rsum[u] = rsum[u] * alpha + ssum;
      rmax[u] = mnew;

      // Store P^T row for this query: lds[q][key], packed 2 x bf16 per dword
      unsigned* row = (unsigned*)(P + (u * 16 + l16) * 32);
#pragma unroll
      for (int i = 0; i < 4; i++) {
        row[(8 * half) / 2 + i]        = pack_bf16(p0[2 * i], p0[2 * i + 1]);
        row[(16 + 8 * half) / 2 + i]   = pack_bf16(p1[2 * i], p1[2 * i + 1]);
      }
      // alpha is uniform per lane -> plain vector x scalar rescale
#pragma unroll
      for (int dt = 0; dt < 4; dt++) o[dt][u] *= alpha;
    }
    asm volatile("s_wait_dscnt 0" ::: "memory");   // LDS stores visible to our reads

    v16bf pf[2];
#pragma unroll
    for (int u = 0; u < 2; u++)
      pf[u] = load_b_frag(P + (u * 16 + l16) * 32, 0, half);  // P^T as B-frag
#pragma unroll
    for (int dt = 0; dt < 4; dt++) {
      v16bf av = load_a_frag(Vp + (size_t)(dt * 16 + l16) * SEQ, kb, half);
      o[dt][0] = WMMA_BF16(av, pf[0], o[dt][0]);
      o[dt][1] = WMMA_BF16(av, pf[1], o[dt][1]);
    }
  }

  // O^T C-layout: N = query = l16, M = dim = dt*16 + r + 8*half -> the 8 dims
  // of one accumulator are contiguous: pack and store 16 bytes at a time.
  const int b = bh >> 4, h = bh & 15;
#pragma unroll
  for (int u = 0; u < 2; u++) {
    float inv = 1.0f / rsum[u];
    int t = q0 + u * 16 + l16;
    bf16_t* outp = O + ((size_t)b * SEQ + t) * CDIM + h * HDIM;
#pragma unroll
    for (int dt = 0; dt < 4; dt++) {
      unsigned pk[4];
#pragma unroll
      for (int i = 0; i < 4; i++)
        pk[i] = pack_bf16(o[dt][u][2 * i] * inv, o[dt][u][2 * i + 1] * inv);
      __builtin_memcpy(outp + dt * 16 + 8 * half, pk, 16);
    }
  }
}

// ---------------------------------------------------------------------------
extern "C" void kernel_launch(void* const* d_in, const int* in_sizes, int n_in,
                              void* d_out, int out_size, void* d_ws, size_t ws_size,
                              hipStream_t stream) {
  const float* x  = (const float*)d_in[0];
  const float* Wq = (const float*)d_in[1];
  const float* bq = (const float*)d_in[2];
  const float* Wk = (const float*)d_in[3];
  const float* bk = (const float*)d_in[4];
  const float* Wv = (const float*)d_in[5];
  const float* bv = (const float*)d_in[6];
  const float* Wo = (const float*)d_in[7];
  const float* bo = (const float*)d_in[8];

  char* ws = (char*)d_ws;
  size_t off = 0;
  bf16_t* xb  = (bf16_t*)(ws + off); off += (size_t)MROWS * CDIM * 2;  // 16 MB
  bf16_t* WqT = (bf16_t*)(ws + off); off += (size_t)CDIM * CDIM * 2;   //  2 MB
  bf16_t* WkT = (bf16_t*)(ws + off); off += (size_t)CDIM * CDIM * 2;
  bf16_t* WvT = (bf16_t*)(ws + off); off += (size_t)CDIM * CDIM * 2;
  bf16_t* WoT = (bf16_t*)(ws + off); off += (size_t)CDIM * CDIM * 2;
  bf16_t* Qb  = (bf16_t*)(ws + off); off += (size_t)MROWS * CDIM * 2;  // 16 MB
  bf16_t* Kb  = (bf16_t*)(ws + off); off += (size_t)MROWS * CDIM * 2;
  bf16_t* Vt  = (bf16_t*)(ws + off); off += (size_t)MROWS * CDIM * 2;
  bf16_t* Ob  = (bf16_t*)(ws + off); off += (size_t)MROWS * CDIM * 2;  // total 88 MB

  const int n_x = MROWS * CDIM;
  convert_kernel<<<n_x / 4 / 256, 256, 0, stream>>>(x, xb, n_x);
  transpose_kernel<<<(CDIM * CDIM) / 256, 256, 0, stream>>>(Wq, WqT);
  transpose_kernel<<<(CDIM * CDIM) / 256, 256, 0, stream>>>(Wk, WkT);
  transpose_kernel<<<(CDIM * CDIM) / 256, 256, 0, stream>>>(Wv, WvT);
  transpose_kernel<<<(CDIM * CDIM) / 256, 256, 0, stream>>>(Wo, WoT);

  dim3 ggrid(CDIM / 256, MROWS / 128);   // (4, 64)
  const float qscale = 0.125f * 1.44269504088896f;  // 1/sqrt(64) * log2(e)
  gemm_kernel<0><<<ggrid, 256, 0, stream>>>(xb, WqT, bq, Qb, qscale);
  gemm_kernel<0><<<ggrid, 256, 0, stream>>>(xb, WkT, bk, Kb, 1.0f);
  gemm_kernel<1><<<ggrid, 256, 0, stream>>>(xb, WvT, bv, Vt, 1.0f);

  attn_kernel<<<dim3(SEQ / 256, BATCH * NHEAD), 256, 0, stream>>>(Qb, Kb, Vt, Ob);

  gemm_kernel<2><<<ggrid, 256, 0, stream>>>(Ob, WoT, bo, d_out, 1.0f);
}
